// GIN_1967095021811
// MI455X (gfx1250) — matrix operand order repaired and checked by reading, not compiled, not run on hardware
//
#include <hip/hip_runtime.h>

// ---------------------------------------------------------------------------
// GIN layer for MI455X (gfx1250): scatter-add aggregation (hw f32 atomics) +
// fused 4-layer MLP using v_wmma_f32_16x16x32_f16 (f16 in, f32 accumulate).
// ---------------------------------------------------------------------------

typedef __attribute__((ext_vector_type(16))) _Float16 v16h;
typedef __attribute__((ext_vector_type(8)))  float    v8f;

#define F_IN      64
#define HID       64
#define HID3      192
#define ACT_PITCH 204        // f32 pitch: 16B-aligned rows, bank-conflict friendly
#define WAVES_PER_BLOCK 4

// ---------------------------------------------------------------------------
// Kernel 1: edge scatter-add.  agg[dst] += x[src]   (agg pre-initialized to x)
// One thread per (edge, float4-of-features): coalesced gather, 4 hw atomics.
// ---------------------------------------------------------------------------
__global__ __launch_bounds__(256)
void gin_edge_scatter(const float* __restrict__ x,
                      const int*   __restrict__ edge_index,
                      float* __restrict__ agg,
                      int E)
{
    size_t tid   = (size_t)blockIdx.x * blockDim.x + threadIdx.x;
    size_t total = (size_t)E * 16;                  // 16 float4s per 64-wide row
    if (tid >= total) return;
    int e = (int)(tid >> 4);
    int q = (int)(tid & 15);
    int s = edge_index[e];                          // row 0 = src
    int d = edge_index[(size_t)E + e];              // row 1 = dst
    const float4 v = *(const float4*)(x + (size_t)s * F_IN + q * 4);
    float* dp = agg + (size_t)d * F_IN + q * 4;
    // agent-scope fp32 atomic add -> global_atomic_add_f32 (return unused)
    (void)__hip_atomic_fetch_add(dp + 0, v.x, __ATOMIC_RELAXED, __HIP_MEMORY_SCOPE_AGENT);
    (void)__hip_atomic_fetch_add(dp + 1, v.y, __ATOMIC_RELAXED, __HIP_MEMORY_SCOPE_AGENT);
    (void)__hip_atomic_fetch_add(dp + 2, v.z, __ATOMIC_RELAXED, __HIP_MEMORY_SCOPE_AGENT);
    (void)__hip_atomic_fetch_add(dp + 3, v.w, __ATOMIC_RELAXED, __HIP_MEMORY_SCOPE_AGENT);
}

// ---------------------------------------------------------------------------
// Kernel 2: weight prep.
//  - fold BN into per-channel scale/shift:  y = (h@W1)*s + b' ; then ReLU
//  - pack W1/W2/Wl1 (f32, [K][N] row-major) into f16 WMMA B-fragments:
//    frag(nt,kb): lane L (<16): n = nt*16+L,    k = kb*32 + j      (j=0..15)
//                 lane L (>=16): n = nt*16+L-16, k = kb*32 + 16 + j
//    stored as [frag][lane][16 halves] -> one aligned 32B load per lane.
// ---------------------------------------------------------------------------
__global__ __launch_bounds__(256)
void gin_prep(const float* __restrict__ W1,  const float* __restrict__ b1,
              const float* __restrict__ gamma, const float* __restrict__ beta,
              const float* __restrict__ mean,  const float* __restrict__ var,
              const float* __restrict__ W2,  const float* __restrict__ Wl1,
              float* __restrict__ bnscale, float* __restrict__ bnbias,
              _Float16* __restrict__ w1f, _Float16* __restrict__ w2f,
              _Float16* __restrict__ wl1f)
{
    int t = blockIdx.x * blockDim.x + threadIdx.x;
    if (t < HID) {
        float s = gamma[t] * rsqrtf(var[t] + 1e-5f);
        bnscale[t] = s;
        bnbias[t]  = (b1[t] - mean[t]) * s + beta[t];
    }
    // 8 frags (W1) + 8 frags (W2) + 24 frags (Wl1); 512 halves per frag
    const int TOTAL = (8 + 8 + 24) * 512;
    for (int idx = t; idx < TOTAL; idx += gridDim.x * blockDim.x) {
        int fid = idx >> 9;
        int rem = idx & 511;
        int ln  = rem >> 4;
        int j   = rem & 15;
        const float* W; _Float16* dst; int f, N;
        if (fid < 8)       { f = fid;      W = W1;  dst = w1f;  N = HID;  }
        else if (fid < 16) { f = fid - 8;  W = W2;  dst = w2f;  N = HID;  }
        else               { f = fid - 16; W = Wl1; dst = wl1f; N = HID3; }
        int nt = f >> 1, kb = f & 1;
        int n = nt * 16 + (ln & 15);
        int k = kb * 32 + ((ln < 16) ? j : 16 + j);
        dst[f * 512 + rem] = (_Float16)W[k * N + n];
    }
}

// ---------------------------------------------------------------------------
// Kernel 3: fused MLP.  One wave32 per 16-node tile.
// LDS activation tile does the D->A transpose between layers (same-wave DS
// ordering, no barriers).  EXEC stays all-1s at every WMMA (uniform early-out).
// ---------------------------------------------------------------------------
__global__ __launch_bounds__(32 * WAVES_PER_BLOCK)
void gin_mlp(const float* __restrict__ agg,
             const float* __restrict__ bnscale, const float* __restrict__ bnbias,
             const _Float16* __restrict__ w1f, const _Float16* __restrict__ w2f,
             const _Float16* __restrict__ wl1f,
             const float* __restrict__ b2,  const float* __restrict__ bl1,
             const float* __restrict__ wl2, const float* __restrict__ bl2,
             float* __restrict__ out, int nNodes, int nTiles)
{
    __shared__ float act_s[WAVES_PER_BLOCK][16][ACT_PITCH];

    const int wave = threadIdx.x >> 5;
    const int lane = threadIdx.x & 31;
    const int tile = blockIdx.x * WAVES_PER_BLOCK + wave;
    if (tile >= nTiles) return;                     // wave-uniform

    float (*A)[ACT_PITCH] = act_s[wave];
    const int base = tile * 16;
    const int m    = lane & 15;                     // A-matrix row (M)
    const int hi   = lane >> 4;
    const int ncol = lane & 15;                     // D-matrix column (N)

    // ---- load 16x64 f32 activation tile (x + agg) into LDS, coalesced -----
    #pragma unroll
    for (int c = 0; c < 8; ++c) {
        int f4  = c * 32 + lane;                    // 256 float4s total
        int row = f4 >> 4;
        int col = (f4 & 15) * 4;
        int node = base + row;
        float4 v = make_float4(0.f, 0.f, 0.f, 0.f);
        if (node < nNodes)
            v = *(const float4*)(agg + (size_t)node * F_IN + col);
        *(float4*)&A[row][col] = v;
    }

    // A-fragment builder: contiguous K order as in the B packing (lane<16: K 0-15;
    // lane>=16: K 16-31), f32 LDS -> f16 convert.
    auto mkfrag = [&](int kb) -> v16h {
        v16h f;
        const int s0 = kb * 32 + (hi ? 16 : 0);
        #pragma unroll
        for (int j = 0; j < 16; ++j) {
            f[j]     = (_Float16)A[m][s0 + j];
        }
        return f;
    };

    // ---- layer 1: h @ W1, BN-affine, ReLU  (K=64, N=64) --------------------
    {
        v16h a0 = mkfrag(0), a1 = mkfrag(1);
        #pragma unroll
        for (int nt = 0; nt < 4; ++nt) {
            v8f c = {};
            v16h bA = *(const v16h*)(w1f + ((size_t)(nt * 2 + 0) * 512 + lane * 16));
            v16h bB = *(const v16h*)(w1f + ((size_t)(nt * 2 + 1) * 512 + lane * 16));
            c = __builtin_amdgcn_wmma_f32_16x16x32_f16(false, a0, false, bA, (short)0, c, false, false);
            c = __builtin_amdgcn_wmma_f32_16x16x32_f16(false, a1, false, bB, (short)0, c, false, false);
            int n = nt * 16 + ncol;
            float s = bnscale[n], bb = bnbias[n];
            #pragma unroll
            for (int r = 0; r < 8; ++r) {
                float v = fmaf(c[r], s, bb);
                A[hi * 8 + r][n] = v > 0.f ? v : 0.f;   // D: lane=N, vgpr=M
            }
        }
    }

    // ---- layer 2: relu(h @ W2 + b2)  (K=64, N=64) --------------------------
    {
        v16h a0 = mkfrag(0), a1 = mkfrag(1);
        #pragma unroll
        for (int nt = 0; nt < 4; ++nt) {
            v8f c = {};
            v16h bA = *(const v16h*)(w2f + ((size_t)(nt * 2 + 0) * 512 + lane * 16));
            v16h bB = *(const v16h*)(w2f + ((size_t)(nt * 2 + 1) * 512 + lane * 16));
            c = __builtin_amdgcn_wmma_f32_16x16x32_f16(false, a0, false, bA, (short)0, c, false, false);
            c = __builtin_amdgcn_wmma_f32_16x16x32_f16(false, a1, false, bB, (short)0, c, false, false);
            int n = nt * 16 + ncol;
            float bb = b2[n];
            #pragma unroll
            for (int r = 0; r < 8; ++r) {
                float v = c[r] + bb;
                A[hi * 8 + r][n] = v > 0.f ? v : 0.f;
            }
        }
    }

    // ---- layer 3: relu(h @ Wl1 + bl1)  (K=64, N=192) -----------------------
    {
        v16h a0 = mkfrag(0), a1 = mkfrag(1);
        #pragma unroll
        for (int nt = 0; nt < 12; ++nt) {
            v8f c = {};
            v16h bA = *(const v16h*)(wl1f + ((size_t)(nt * 2 + 0) * 512 + lane * 16));
            v16h bB = *(const v16h*)(wl1f + ((size_t)(nt * 2 + 1) * 512 + lane * 16));
            c = __builtin_amdgcn_wmma_f32_16x16x32_f16(false, a0, false, bA, (short)0, c, false, false);
            c = __builtin_amdgcn_wmma_f32_16x16x32_f16(false, a1, false, bB, (short)0, c, false, false);
            int n = nt * 16 + ncol;
            float bb = bl1[n];
            #pragma unroll
            for (int r = 0; r < 8; ++r) {
                float v = c[r] + bb;
                A[hi * 8 + r][n] = v > 0.f ? v : 0.f;
            }
        }
    }

    // ---- head: out = act(16x192) . Wl2 + bl2  (2 lanes per node) ----------
    {
        int node = lane >> 1;
        int half = lane & 1;
        float sum = 0.f;
        const int k0 = half * 96;
        #pragma unroll 8
        for (int k = 0; k < 96; ++k)
            sum = fmaf(A[node][k0 + k], wl2[k0 + k], sum);
        sum += __shfl_xor(sum, 1);
        if (half == 0 && (base + node) < nNodes)
            out[base + node] = sum + bl2[0];
    }
}

// ---------------------------------------------------------------------------
extern "C" void kernel_launch(void* const* d_in, const int* in_sizes, int n_in,
                              void* d_out, int out_size, void* d_ws, size_t ws_size,
                              hipStream_t stream) {
    const float* x     = (const float*)d_in[0];
    const int*   ei    = (const int*)  d_in[1];
    const float* W1    = (const float*)d_in[2];
    const float* b1    = (const float*)d_in[3];
    const float* gamma = (const float*)d_in[4];
    const float* beta  = (const float*)d_in[5];
    const float* mean  = (const float*)d_in[6];
    const float* var   = (const float*)d_in[7];
    const float* W2    = (const float*)d_in[8];
    const float* b2    = (const float*)d_in[9];
    const float* Wl1   = (const float*)d_in[10];
    const float* bl1   = (const float*)d_in[11];
    const float* Wl2   = (const float*)d_in[12];
    const float* bl2   = (const float*)d_in[13];

    const int N = in_sizes[0] / F_IN;       // 100000
    const int E = in_sizes[1] / 2;          // 3200000

    // workspace layout (all offsets 256B aligned)
    char* ws = (char*)d_ws;
    size_t off = 0;
    float* agg = (float*)(ws + off);            off += (size_t)N * F_IN * sizeof(float);
    off = (off + 255) & ~(size_t)255;
    float* bnscale = (float*)(ws + off);        off += 256;
    float* bnbias  = (float*)(ws + off);        off += 256;
    _Float16* w1f  = (_Float16*)(ws + off);     off += 8  * 512 * sizeof(_Float16); off = (off + 255) & ~(size_t)255;
    _Float16* w2f  = (_Float16*)(ws + off);     off += 8  * 512 * sizeof(_Float16); off = (off + 255) & ~(size_t)255;
    _Float16* wl1f = (_Float16*)(ws + off);     off += 24 * 512 * sizeof(_Float16);
    (void)ws_size; (void)n_in; (void)out_size;

    // agg := x  (identity term of GIN, eps = 0)
    hipMemcpyAsync(agg, x, (size_t)N * F_IN * sizeof(float),
                   hipMemcpyDeviceToDevice, stream);

    // weight packing + BN folding (tiny)
    gin_prep<<<80, 256, 0, stream>>>(W1, b1, gamma, beta, mean, var, W2, Wl1,
                                     bnscale, bnbias, w1f, w2f, wl1f);

    // scatter-add over edges
    {
        size_t total  = (size_t)E * 16;
        int    blocks = (int)((total + 255) / 256);
        gin_edge_scatter<<<blocks, 256, 0, stream>>>(x, ei, agg, E);
    }

    // fused MLP (WMMA)
    {
        int nTiles = (N + 15) / 16;
        int blocks = (nTiles + WAVES_PER_BLOCK - 1) / WAVES_PER_BLOCK;
        gin_mlp<<<blocks, 32 * WAVES_PER_BLOCK, 0, stream>>>(
            agg, bnscale, bnbias, w1f, w2f, wl1f,
            b2, bl1, Wl2, bl2, (float*)d_out, N, nTiles);
    }
}
